// privacyLoss_63496796504179
// MI455X (gfx1250) — compile-verified, matrix operand order checked by reading, not run on hardware
//
#include <hip/hip_runtime.h>
#include <hip/hip_bf16.h>

typedef __attribute__((ext_vector_type(16))) _Float16 v16h;
typedef __attribute__((ext_vector_type(8)))  float    v8f;

#define N_ROWS 8192
#define N_DIM  256
#define N_CLS  10
#define NBLK_A 64
#define ROWS_PER_BLK  (N_ROWS / NBLK_A)    // 128
#define ROWS_PER_WAVE (ROWS_PER_BLK / 8)   // 16
#define PAR_CONST (0.5 / 8192.0)

// ---------------------------------------------------------------------------
// Pass A: stream the 8 MB input once. Per row: squared L2 norm (wave32 shuffle
// reduction) and per-class column-sum accumulation into per-wave LDS copies
// (8 waves x 10 classes x 256 cols x f32 = 80 KB of the 320 KB WGP LDS), so no
// atomics anywhere. Per-block partial class sums go to workspace.
// ---------------------------------------------------------------------------
__global__ __launch_bounds__(256) void pl_passA(const float* __restrict__ x,
                                                const int*   __restrict__ label,
                                                float* __restrict__ partialCls, // [NBLK_A][10*256]
                                                float* __restrict__ sqArr) {    // [8192]
  __shared__ float wcls[8 * N_CLS * N_DIM]; // per-wave accumulators, 80 KB
  const int tid  = threadIdx.x;
  const int wave = tid >> 5;
  const int lane = tid & 31;

  for (int i = tid; i < 8 * N_CLS * N_DIM; i += 256) wcls[i] = 0.0f;
  __syncthreads();

  float* mycls = &wcls[wave * N_CLS * N_DIM];
  const int rowBase = blockIdx.x * ROWS_PER_BLK + wave * ROWS_PER_WAVE;

  for (int r = 0; r < ROWS_PER_WAVE; ++r) {
    const int row = rowBase + r;
    const int c   = label[row];
    const float4* rp = (const float4*)(x + (size_t)row * N_DIM);
    float4 v0 = rp[lane];        // cols 4*lane   .. 4*lane+3
    float4 v1 = rp[lane + 32];   // cols 128+4*lane .. 131+4*lane

    float s = v0.x*v0.x + v0.y*v0.y + v0.z*v0.z + v0.w*v0.w
            + v1.x*v1.x + v1.y*v1.y + v1.z*v1.z + v1.w*v1.w;
    #pragma unroll
    for (int off = 16; off > 0; off >>= 1) s += __shfl_xor(s, off, 32);
    if (lane == 0) sqArr[row] = s;

    float* d0 = &mycls[c * N_DIM + 4 * lane];  // lanes own disjoint columns
    d0[0] += v0.x; d0[1] += v0.y; d0[2] += v0.z; d0[3] += v0.w;
    float* d1 = d0 + 128;
    d1[0] += v1.x; d1[1] += v1.y; d1[2] += v1.z; d1[3] += v1.w;
  }
  __syncthreads();

  for (int idx = tid; idx < N_CLS * N_DIM; idx += 256) {
    float a = 0.0f;
    #pragma unroll
    for (int w = 0; w < 8; ++w) a += wcls[w * N_CLS * N_DIM + idx];
    partialCls[(size_t)blockIdx.x * (N_CLS * N_DIM) + idx] = a;
  }
}

// ---------------------------------------------------------------------------
// Pass B (single block): reduce partials, build corrected S^R / S^C matrices
// (classes 0..9, row 10 = total, rows 11..15 zero-padded), run the 16x16x256
// class Gram with chained v_wmma_f32_16x16x32_f16 on wave 0, and finish the
// weighted-squared-norm sum in double precision.
// ---------------------------------------------------------------------------
__global__ __launch_bounds__(256) void pl_passB(const float* __restrict__ x,
                                                const int*   __restrict__ label,
                                                const float* __restrict__ partialCls,
                                                const float* __restrict__ sqArr,
                                                float* __restrict__ result) {
  __shared__ float    cls[N_CLS * N_DIM];  // reduced class sums (10 KB)
  __shared__ _Float16 SRh[16 * N_DIM];     // A matrix rows (8 KB)
  __shared__ _Float16 SCh[16 * N_DIM];     // B matrix cols (8 KB)
  __shared__ int      cnt[16];
  __shared__ float    diag[16];
  __shared__ double   partial[256];

  const int tid = threadIdx.x;
  const int N = N_ROWS;

  if (tid < 16) cnt[tid] = 0;
  __syncthreads();

  for (int i = tid; i < N; i += 256) atomicAdd(&cnt[label[i]], 1);

  for (int idx = tid; idx < N_CLS * N_DIM; idx += 256) {
    float a = 0.0f;
    for (int b = 0; b < NBLK_A; ++b) a += partialCls[(size_t)b * (N_CLS * N_DIM) + idx];
    cls[idx] = a;
  }
  __syncthreads();

  const int l0 = label[0];
  const int lN = label[N - 1];

  // S^R_c = cls_c - [l_{N-1}==c] x_{N-1};  S^C_c = cls_c - [l_0==c] x_0
  // row 10 = totals with the same corrections; rows 11..15 zero.
  for (int idx = tid; idx < 16 * N_DIM; idx += 256) {
    const int c = idx >> 8, d = idx & 255;
    const float x0 = x[d];
    const float xN = x[(size_t)(N - 1) * N_DIM + d];
    float sr, sc;
    if (c < N_CLS) {
      const float base = cls[c * N_DIM + d];
      sr = base - ((lN == c) ? xN : 0.0f);
      sc = base - ((l0 == c) ? x0 : 0.0f);
    } else if (c == 10) {
      float tot = 0.0f;
      #pragma unroll
      for (int k = 0; k < N_CLS; ++k) tot += cls[k * N_DIM + d];
      sr = tot - xN;
      sc = tot - x0;
    } else {
      sr = 0.0f; sc = 0.0f;
    }
    SRh[idx] = (_Float16)sr;
    SCh[idx] = (_Float16)sc;
  }
  __syncthreads();

  // Class Gram D = SR(16x256) x SC^T(256x16) on wave 0 (EXEC all ones).
  if (tid < 32) {
    const int lane = tid;
    const int m    = lane & 15;
    const int hi   = lane >> 4;
    v8f acc = {};
    for (int step = 0; step < 8; ++step) {
      const int k0 = step * 32;
      v16h a, b;
      #pragma unroll
      for (int h = 0; h < 16; ++h) {
        // 16-bit A 16x32 layout: halves 0..7 -> K=0..7 (+8 for hi lanes),
        //                        halves 8..15 -> K=16..23 (+8 for hi lanes)
        const int K = k0 + ((h < 8) ? h : h + 8) + 8 * hi;
        a[h] = SRh[m * N_DIM + K];
        b[h] = SCh[m * N_DIM + K];   // B = SC^T: N index == lane&15, same K packing
      }
      acc = __builtin_amdgcn_wmma_f32_16x16x32_f16(
          /*neg_a=*/false, a, /*neg_b=*/false, b,
          /*c_mod=*/(short)0, acc, /*reuse_a=*/false, /*reuse_b=*/false);
    }
    float tmp[8];
    #pragma unroll
    for (int i = 0; i < 8; ++i) tmp[i] = acc[i];
    // D layout: lane (hi,m) holds D[v + 8*hi][m] in component v.
    if (hi == 0 && m < 8)  diag[m] = tmp[m];        // D[c][c], c = 0..7
    if (hi == 1 && m >= 8) diag[m] = tmp[m - 8];    // D[c][c], c = 8..15
  }
  __syncthreads();

  // Part1: weighted sum of squared norms (double accumulation).
  double p = 0.0;
  for (int i = tid; i < N; i += 256) {
    const int li = label[i];
    const float s = sqArr[i];
    const int cC = cnt[li] - ((l0 == li) ? 1 : 0);  // class count over C = {1..N-1}
    const int cR = cnt[li] - ((lN == li) ? 1 : 0);  // class count over R = {0..N-2}
    double w = 0.0;
    if (i <= N - 2) w += (double)(2 * cC - (N - 1));
    if (i >= 1)     w += (double)(2 * cR - (N - 1));
    p += (double)s * w;
  }
  partial[tid] = p;
  __syncthreads();

  if (tid == 0) {
    double P1 = 0.0;
    for (int t = 0; t < 256; ++t) P1 += partial[t];
    double P2 = 0.0;
    for (int c = 0; c < N_CLS; ++c) P2 += (double)diag[c];
    P2 = 2.0 * P2 - (double)diag[10];
    const double total = P1 - 2.0 * P2;
    result[0] = (float)(PAR_CONST * total);
  }
}

extern "C" void kernel_launch(void* const* d_in, const int* in_sizes, int n_in,
                              void* d_out, int out_size, void* d_ws, size_t ws_size,
                              hipStream_t stream) {
  (void)in_sizes; (void)n_in; (void)out_size; (void)ws_size;
  const float* x     = (const float*)d_in[0];
  const int*   label = (const int*)d_in[1];

  float* partialCls = (float*)d_ws;                          // 64 * 2560 floats
  float* sqArr      = partialCls + NBLK_A * N_CLS * N_DIM;   // 8192 floats

  pl_passA<<<NBLK_A, 256, 0, stream>>>(x, label, partialCls, sqArr);
  pl_passB<<<1, 256, 0, stream>>>(x, label, partialCls, sqArr, (float*)d_out);
}
